// BasicAttention_50869592654505
// MI455X (gfx1250) — matrix operand order repaired and checked by reading, not compiled, not run on hardware
//
#include <hip/hip_runtime.h>
#include <hip/hip_bf16.h>
#include <math.h>

// ---------------------------------------------------------------------------
// BasicAttention (Bahdanau-style additive attention), B=32, S=2048, D=1024.
// Dominant cost: source = context @ W_c.T (137 GFLOP) -> bf16 WMMA, fp32 acc,
// fused with tanh/score epilogue so the [B,S,D] "source" tensor never exists.
// v2: 2 M-tiles per wave so each B fragment feeds two WMMAs (halves L2
// traffic for W_c, the binding constraint once the GEMM is XDL-bound).
// ---------------------------------------------------------------------------

#define BATCH 32
#define SEQ   2048
#define DIM   1024
#define KC    512      // K-chunk staged in LDS (32 rows x 512 x bf16 = 32 KB)

typedef __attribute__((ext_vector_type(16))) __bf16          v16bf;
typedef __attribute__((ext_vector_type(8)))  float           v8f;
typedef __attribute__((ext_vector_type(8)))  unsigned short  u16x8;

union Frag16 {
    v16bf bf;
    u16x8 h[2];     // h[0] = elements 0..7 (VGPR 0..3), h[1] = elements 8..15 (VGPR 4..7)
};

__device__ __forceinline__ unsigned short f32_to_bf16_rne(float f) {
    unsigned int u = __float_as_uint(f);
    u += 0x7FFFu + ((u >> 16) & 1u);       // round-to-nearest-even
    return (unsigned short)(u >> 16);
}

// ---------------------------------------------------------------------------
// Kernel 1: W_c f32 -> bf16 (done once; 2 MB result stays L2-resident)
// ---------------------------------------------------------------------------
__global__ __launch_bounds__(256) void convert_bf16_kernel(
    const float* __restrict__ in, unsigned short* __restrict__ out, int n)
{
    int i = blockIdx.x * 256 + threadIdx.x;
    if (i < n) out[i] = f32_to_bf16_rne(in[i]);
}

// ---------------------------------------------------------------------------
// Kernel 2: generic fp32 row-dot with block reduction:
//   out[b,e] = sum_j A[b,j]*W[e,j] + bias[e]
// Used for target = x @ W_in.T + b_c
// ---------------------------------------------------------------------------
__global__ __launch_bounds__(256) void rowdot_kernel(
    const float* __restrict__ A, const float* __restrict__ W,
    const float* __restrict__ bias, float* __restrict__ out,
    int K, int N, int do_tanh)
{
    __shared__ float red[256];
    int e = blockIdx.x % N;
    int b = blockIdx.x / N;
    const float* a = A + (size_t)b * K;
    const float* w = W + (size_t)e * K;
    float s = 0.f;
    for (int j = threadIdx.x; j < K; j += 256) s += a[j] * w[j];
    red[threadIdx.x] = s;
    __syncthreads();
    for (int o = 128; o > 0; o >>= 1) {
        if (threadIdx.x < o) red[threadIdx.x] += red[threadIdx.x + o];
        __syncthreads();
    }
    if (threadIdx.x == 0) {
        float v = red[0] + (bias ? bias[e] : 0.f);
        out[(size_t)b * N + e] = do_tanh ? tanhf(v) : v;
    }
}

// ---------------------------------------------------------------------------
// Kernel 3 (hot): scores[r] = sum_e tanh(ctx_row_r @ W_c.T + T[b])_e * w_v[e]
//   - grid.x = (B*S)/32 row tiles, 256 threads (8 waves)
//   - A panel: 32 rows x 512-K chunk of context as bf16 in 32 KB LDS,
//     re-staged once (K=1024 in two chunks); context read once from HBM
//   - each wave: 2 M-tiles x 8 N-tiles -> 16 WMMAs per k-step, each B
//     fragment reused by both M-tiles (128 accumulator VGPRs)
// VGPR layouts per CDNA5 ISA 7.12.2:
//   A (16x32, 16-bit): lane m(0-15) holds M=m, K={k0..k0+7, k0+16..k0+23};
//                      lane m+16 holds K={k0+8..k0+15, k0+24..k0+31}
//   B (32x16, 16-bit): lane n(0-15) holds N=n, K={k0..k0+7, k0+16..k0+23};
//                      lane n+16 holds K={k0+8..k0+15, k0+24..k0+31} (same as A)
//   C/D (16x16 f32):   lane holds N=lane&15, VGPR i -> M = 8*(lane>>4) + i
// ---------------------------------------------------------------------------
__global__ __launch_bounds__(256) void fused_scores_kernel(
    const float*          __restrict__ ctx,     // [B*S, D] fp32
    const unsigned short* __restrict__ Wc_bf,   // [D, D] bf16 (row e, col d)
    const float*          __restrict__ T,       // [B, D] = x@W_in.T + b_c
    const float*          __restrict__ w_v,     // [D]
    float*                __restrict__ scores)  // [B*S]
{
    __shared__ unsigned short Abuf[32 * KC];    // 32 KB
    __shared__ float          sc[32];

    const int tid  = threadIdx.x;
    const int wave = tid >> 5;       // 0..7  -> owns N-tiles wave*8 .. wave*8+7
    const int lane = tid & 31;
    const int r0   = blockIdx.x * 32;           // first flattened (b,s) row
    const int b    = r0 / SEQ;                  // 32-row tile never crosses batches

    if (tid < 32) sc[tid] = 0.f;

    const int m = lane & 15;
    const int h = lane >> 4;

    v8f acc[2][8];
    const v8f vz = {0.f, 0.f, 0.f, 0.f, 0.f, 0.f, 0.f, 0.f};
#pragma unroll
    for (int mt = 0; mt < 2; ++mt)
#pragma unroll
        for (int n = 0; n < 8; ++n) acc[mt][n] = vz;

    for (int kc = 0; kc < DIM; kc += KC) {
        __syncthreads();    // Abuf safe to overwrite (also orders sc init)
        // Stage A panel: 32 rows x KC cols, fp32 -> bf16 (coalesced)
        for (int i = tid; i < 32 * KC; i += 256) {
            const int row = i >> 9;             // / KC
            const int col = i & (KC - 1);       // % KC
            Abuf[i] = f32_to_bf16_rne(ctx[(size_t)(r0 + row) * DIM + kc + col]);
        }
        __syncthreads();

        for (int k0 = 0; k0 < KC; k0 += 32) {
            // Two A fragments (M-tiles 0 and 1) from LDS
            Frag16 a0, a1;
            const unsigned short* ab = &Abuf[m * KC + k0 + 8 * h];
            a0.h[0] = *(const u16x8*)(ab);
            a0.h[1] = *(const u16x8*)(ab + 16);
            const unsigned short* ab1 = ab + 16 * KC;
            a1.h[0] = *(const u16x8*)(ab1);
            a1.h[1] = *(const u16x8*)(ab1 + 16);
#pragma unroll
            for (int n = 0; n < 8; ++n) {
                const int e = (wave * 8 + n) * 16 + m;   // B column for this lane
                Frag16 bf;
                const unsigned short* wb = Wc_bf + (size_t)e * DIM + kc + k0 + 8 * h;
                bf.h[0] = *(const u16x8*)(wb);
                bf.h[1] = *(const u16x8*)(wb + 16);
                acc[0][n] = __builtin_amdgcn_wmma_f32_16x16x32_bf16(
                    false, a0.bf, false, bf.bf, (short)0, acc[0][n], false, false);
                acc[1][n] = __builtin_amdgcn_wmma_f32_16x16x32_bf16(
                    false, a1.bf, false, bf.bf, (short)0, acc[1][n], false, false);
            }
        }
    }

    // Fused epilogue: tanh(source + target + b_c) dotted with w_v, per row
    float p[2][8];
#pragma unroll
    for (int mt = 0; mt < 2; ++mt)
#pragma unroll
        for (int i = 0; i < 8; ++i) p[mt][i] = 0.f;

#pragma unroll
    for (int n = 0; n < 8; ++n) {
        const int e  = (wave * 8 + n) * 16 + m;         // N column = lane&15
        const float te = T[(size_t)b * DIM + e];
        const float wv = w_v[e];
#pragma unroll
        for (int i = 0; i < 8; ++i) {
            p[0][i] += tanhf(acc[0][n][i] + te) * wv;   // rows r0 +      8h+i
            p[1][i] += tanhf(acc[1][n][i] + te) * wv;   // rows r0 + 16 + 8h+i
        }
    }
#pragma unroll
    for (int mt = 0; mt < 2; ++mt)
#pragma unroll
        for (int i = 0; i < 8; ++i)
            atomicAdd(&sc[mt * 16 + 8 * h + i], p[mt][i]);   // ds_add_f32
    __syncthreads();
    if (tid < 32) scores[(size_t)r0 + tid] = sc[tid];
}

// ---------------------------------------------------------------------------
// Kernel 4: softmax over S per batch row -> writes attn_w directly into d_out
// ---------------------------------------------------------------------------
__global__ __launch_bounds__(256) void softmax_kernel(
    const float* __restrict__ scores, float* __restrict__ attn)
{
    __shared__ float red[256];
    __shared__ float sM, sSum;
    const int b = blockIdx.x, t = threadIdx.x;
    const float* sr = scores + (size_t)b * SEQ;

    float m = -3.4e38f;
    for (int s = t; s < SEQ; s += 256) m = fmaxf(m, sr[s]);
    red[t] = m; __syncthreads();
    for (int o = 128; o > 0; o >>= 1) {
        if (t < o) red[t] = fmaxf(red[t], red[t + o]);
        __syncthreads();
    }
    if (t == 0) sM = red[0];
    __syncthreads();
    const float gm = sM;

    float sum = 0.f;
    for (int s = t; s < SEQ; s += 256) sum += __expf(sr[s] - gm);
    red[t] = sum; __syncthreads();
    for (int o = 128; o > 0; o >>= 1) {
        if (t < o) red[t] += red[t + o];
        __syncthreads();
    }
    if (t == 0) sSum = red[0];
    __syncthreads();
    const float inv = 1.f / sSum;

    for (int s = t; s < SEQ; s += 256)
        attn[(size_t)b * SEQ + s] = __expf(sr[s] - gm) * inv;
}

// ---------------------------------------------------------------------------
// Kernel 5: weighted[b,e] = sum_s attn_w[b,s] * context[b,s,e]
// Coalesced across e; attn_w[b,s] is wave-uniform -> scalar loads.
// ---------------------------------------------------------------------------
__global__ __launch_bounds__(256) void weighted_kernel(
    const float* __restrict__ attn, const float* __restrict__ ctx,
    float* __restrict__ weighted)
{
    const int b = blockIdx.y;
    const int e = blockIdx.x * 256 + threadIdx.x;
    const float* c  = ctx + ((size_t)b * SEQ) * DIM + e;
    const float* aw = attn + (size_t)b * SEQ;
    float acc = 0.f;
    for (int s = 0; s < SEQ; ++s)
        acc += aw[s] * c[(size_t)s * DIM];
    weighted[(size_t)b * DIM + e] = acc;
}

// ---------------------------------------------------------------------------
// Kernel 6: h_tilde[b,e] = tanh( [weighted[b,:], x[b,:]] @ W_out[e,:] )
// ---------------------------------------------------------------------------
__global__ __launch_bounds__(256) void final_kernel(
    const float* __restrict__ weighted, const float* __restrict__ x,
    const float* __restrict__ Wout, float* __restrict__ out)
{
    __shared__ float red[256];
    const int e = blockIdx.x & (DIM - 1);
    const int b = blockIdx.x >> 10;
    const float* w = Wout + (size_t)e * (2 * DIM);
    float s = 0.f;
    for (int j = threadIdx.x; j < 2 * DIM; j += 256) {
        float c = (j < DIM) ? weighted[(size_t)b * DIM + j]
                            : x[(size_t)b * DIM + (j - DIM)];
        s += c * w[j];
    }
    red[threadIdx.x] = s; __syncthreads();
    for (int o = 128; o > 0; o >>= 1) {
        if (threadIdx.x < o) red[threadIdx.x] += red[threadIdx.x + o];
        __syncthreads();
    }
    if (threadIdx.x == 0)
        out[(size_t)b * DIM + e] = tanhf(red[0]);
}

// ---------------------------------------------------------------------------
extern "C" void kernel_launch(void* const* d_in, const int* in_sizes, int n_in,
                              void* d_out, int out_size, void* d_ws, size_t ws_size,
                              hipStream_t stream) {
    const float* x    = (const float*)d_in[0];   // [32,1024]
    const float* ctx  = (const float*)d_in[1];   // [32,2048,1024]
    const float* Win  = (const float*)d_in[2];   // [1024,1024]
    const float* Wc   = (const float*)d_in[3];   // [1024,1024]
    const float* bc   = (const float*)d_in[4];   // [1024]
    const float* wv   = (const float*)d_in[5];   // [1024]
    const float* Wout = (const float*)d_in[6];   // [1024,2048]

    float* out_h    = (float*)d_out;                        // h_tilde [32,1024]
    float* out_attn = (float*)d_out + BATCH * DIM;          // attn_w  [32,2048]

    // Workspace layout (2.5 MB total)
    char* ws = (char*)d_ws;
    unsigned short* Wc_bf = (unsigned short*)ws;                    // 2 MB
    float* T        = (float*)(ws + (2u << 20));                    // 128 KB
    float* scores   = (float*)(ws + (2u << 20) + (128u << 10));     // 256 KB
    float* weighted = (float*)(ws + (2u << 20) + (384u << 10));     // 128 KB

    // 1) W_c -> bf16 (L2-resident for the whole GEMM)
    convert_bf16_kernel<<<(DIM * DIM) / 256, 256, 0, stream>>>(Wc, Wc_bf, DIM * DIM);

    // 2) T = x @ W_in.T + b_c   (fp32, tiny)
    rowdot_kernel<<<BATCH * DIM, 256, 0, stream>>>(x, Win, bc, T, DIM, DIM, 0);

    // 3) Fused WMMA GEMM + tanh + w_v dot -> scores
    fused_scores_kernel<<<(BATCH * SEQ) / 32, 256, 0, stream>>>(ctx, Wc_bf, T, wv, scores);

    // 4) softmax -> attn_w (second output)
    softmax_kernel<<<BATCH, 256, 0, stream>>>(scores, out_attn);

    // 5) weighted = attn_w @ context  (fp32, HBM-bound second pass over context)
    weighted_kernel<<<dim3(DIM / 256, BATCH), 256, 0, stream>>>(out_attn, ctx, weighted);

    // 6) h_tilde (first output)
    final_kernel<<<BATCH * DIM, 256, 0, stream>>>(weighted, x, Wout, out_h);
}
